// cLSTM_15633680957682
// MI455X (gfx1250) — compile-verified, matrix operand order checked by reading, not run on hardware
//
#include <hip/hip_runtime.h>
#include <hip/hip_bf16.h>
#include <math.h>

typedef __attribute__((ext_vector_type(16))) _Float16 v16h;
typedef __attribute__((ext_vector_type(8)))  _Float16 v8h;
typedef __attribute__((ext_vector_type(8)))  float    v8f;

#define B_DIM 16384
#define IN_DIM 1024
#define H_DIM 1024
#define KRED 1024              // effective K (h==0 kills bottom half of W)
#define NGATES 5               // i, it, z, o, d  (f / f_target are dead: c == c_target == 0)
#define MT 2                   // M-tiles per wave (32 rows) sharing B fragments
#define BH ((size_t)B_DIM * H_DIM)
#define GATE_STRIDE (1u << 20) // 64 n16 * 32 k32 * 512 halves per gate

union V16U { v16h v; v8h h[2]; };

// ---------------- pack kernels ----------------

// x (f32, [B, 1024] row-major) -> xh (f16, same layout)
__global__ void clstm_pack_x(const float* __restrict__ x, _Float16* __restrict__ xh) {
    size_t i = ((size_t)blockIdx.x * blockDim.x + threadIdx.x) * 4;
    float4 v = *(const float4*)(x + i);
    xh[i + 0] = (_Float16)v.x;
    xh[i + 1] = (_Float16)v.y;
    xh[i + 2] = (_Float16)v.z;
    xh[i + 3] = (_Float16)v.w;
}

// Pack top 1024 rows of each live W (f32, [2048,1024] row-major) into WMMA
// B-fragment layout: idx = g<<20 | n16<<14 | k32<<9 | lane<<4 | e
// lane 0..15 -> N = n16*16+lane, K base 0;  lane 16..31 -> same N set, K base 8
// element e<8 -> K = base+e ; e>=8 -> K = 16+base+(e-8)
__global__ void clstm_pack_w(_Float16* __restrict__ wp,
                             const float* __restrict__ W0, const float* __restrict__ W1,
                             const float* __restrict__ W2, const float* __restrict__ W3,
                             const float* __restrict__ W4) {
    unsigned idx = blockIdx.x * blockDim.x + threadIdx.x;   // 0 .. 5*2^20-1
    unsigned e    = idx & 15;
    unsigned lane = (idx >> 4) & 31;
    unsigned k32  = (idx >> 9) & 31;
    unsigned n16  = (idx >> 14) & 63;
    unsigned g    = idx >> 20;
    unsigned base = (lane & 16) ? 8u : 0u;
    unsigned kl   = (e < 8) ? (base + e) : (16u + base + (e - 8));
    unsigned k    = k32 * 32 + kl;
    unsigned n    = n16 * 16 + (lane & 15);
    const float* W = (g == 0) ? W0 : (g == 1) ? W1 : (g == 2) ? W2 : (g == 3) ? W3 : W4;
    wp[idx] = (_Float16)W[(size_t)k * H_DIM + n];
}

// ---------------- fused GEMM + epilogue ----------------

__device__ __forceinline__ float sigmoidf_(float v) { return 1.0f / (1.0f + __expf(-v)); }

__global__ void __launch_bounds__(128)
clstm_gemm(const _Float16* __restrict__ xh, const _Float16* __restrict__ wp,
           const float* __restrict__ bi, const float* __restrict__ bit,
           const float* __restrict__ bz, const float* __restrict__ bo,
           const float* __restrict__ bd,
           const float* __restrict__ t, const float* __restrict__ last_t,
           float* __restrict__ out) {
    const int lane  = threadIdx.x & 31;
    const int wave  = threadIdx.x >> 5;
    const int m0    = blockIdx.x * 128 + wave * 32;      // 32-row slab per wave
    const int n16   = blockIdx.y;                        // 16-col tile index
    const int mrow  = m0 + (lane & 15);
    const int kbase = (lane & 16) ? 8 : 0;

    const _Float16* pA0 = xh + (size_t)mrow * KRED + kbase;              // m-tile 0
    const _Float16* pA1 = pA0 + (size_t)16 * KRED;                       // m-tile 1
    const _Float16* pB  = wp + (size_t)n16 * 32 * 512 + (size_t)lane * 16;

    v8f acc[MT][NGATES];
    #pragma unroll
    for (int mt = 0; mt < MT; ++mt)
        #pragma unroll
        for (int g = 0; g < NGATES; ++g) acc[mt][g] = (v8f){};

    // Double-buffered fragments: loads for step kk+1 overlap WMMAs of step kk.
    // B fragments are shared by both M-tiles -> ~23 FLOP/byte from L2.
    V16U a[2][MT];
    v16h b[2][NGATES];

    a[0][0].h[0] = *(const v8h*)(pA0);
    a[0][0].h[1] = *(const v8h*)(pA0 + 16);
    a[0][1].h[0] = *(const v8h*)(pA1);
    a[0][1].h[1] = *(const v8h*)(pA1 + 16);
    #pragma unroll
    for (int g = 0; g < NGATES; ++g)
        b[0][g] = *(const v16h*)(pB + (size_t)g * GATE_STRIDE);

    #pragma unroll 2
    for (int kk = 0; kk < KRED / 32; ++kk) {
        const int cur = kk & 1;
        const int nxt = cur ^ 1;
        const int kn  = (kk + 1) & 31;                   // branchless wrap; extra wrapped
                                                         // load on last step is harmless
        const _Float16* pa0 = pA0 + (size_t)kn * 32;
        const _Float16* pa1 = pA1 + (size_t)kn * 32;
        a[nxt][0].h[0] = *(const v8h*)(pa0);
        a[nxt][0].h[1] = *(const v8h*)(pa0 + 16);
        a[nxt][1].h[0] = *(const v8h*)(pa1);
        a[nxt][1].h[1] = *(const v8h*)(pa1 + 16);
        const _Float16* pk = pB + (size_t)kn * 512;
        #pragma unroll
        for (int g = 0; g < NGATES; ++g)
            b[nxt][g] = *(const v16h*)(pk + (size_t)g * GATE_STRIDE);

        #pragma unroll
        for (int g = 0; g < NGATES; ++g) {
            acc[0][g] = __builtin_amdgcn_wmma_f32_16x16x32_f16(
                false, a[cur][0].v, false, b[cur][g], (short)0, acc[0][g], false, false);
            acc[1][g] = __builtin_amdgcn_wmma_f32_16x16x32_f16(
                false, a[cur][1].v, false, b[cur][g], (short)0, acc[1][g], false, false);
        }
    }

    // ---- fused epilogue ----
    const int n = n16 * 16 + (lane & 15);
    const float bI  = bi[n];
    const float bIT = bit[n];
    const float bZ  = bz[n];
    const float bO  = bo[n];
    const float bD  = bd[n];

    float* oh  = out;                 // h_n
    float* oc  = out + BH;            // c_n
    float* oct = out + 2 * BH;        // c_target_n
    float* oo  = out + 3 * BH;        // gate_o
    float* od  = out + 4 * BH;        // delta

    #pragma unroll
    for (int mt = 0; mt < MT; ++mt) {
        const int rowbase = m0 + mt * 16 + ((lane & 16) ? 8 : 0);
        #pragma unroll
        for (int r = 0; r < 8; ++r) {
            const int row = rowbase + r;
            const float gi  = acc[mt][0][r] + bI;
            const float git = acc[mt][1][r] + bIT;
            const float gz  = acc[mt][2][r] + bZ;
            const float go  = acc[mt][3][r] + bO;
            const float gd  = acc[mt][4][r] + bD;

            const float z     = tanhf(0.5f * gz);          // 2*sigmoid(gz) - 1
            const float si    = sigmoidf_(gi);
            const float sit   = sigmoidf_(git);
            const float so    = sigmoidf_(go);
            const float c_n   = si * z;                    // gate_f * 0 dropped
            const float ct_n  = sit * z;
            const float delta = fmaxf(gd, 0.0f) + log1pf(__expf(-fabsf(gd)));
            const float dt    = t[row] - last_t[row];
            const float cell  = ct_n + (c_n - ct_n) * __expf(-delta * dt);
            const float h_n   = so * tanhf(cell);

            const size_t o = (size_t)row * H_DIM + n;
            oh[o]  = h_n;
            oc[o]  = c_n;
            oct[o] = ct_n;
            oo[o]  = so;
            od[o]  = delta;
        }
    }
}

// ---------------- launcher ----------------

extern "C" void kernel_launch(void* const* d_in, const int* in_sizes, int n_in,
                              void* d_out, int out_size, void* d_ws, size_t ws_size,
                              hipStream_t stream) {
    (void)in_sizes; (void)n_in; (void)out_size; (void)ws_size;
    const float* x      = (const float*)d_in[0];
    // d_in[1] = n_id (unused by the reference math)
    const float* t      = (const float*)d_in[2];
    const float* last_t = (const float*)d_in[3];
    const float* Wi  = (const float*)d_in[4];   const float* bi  = (const float*)d_in[5];
    const float* Wit = (const float*)d_in[6];   const float* bit = (const float*)d_in[7];
    // d_in[8..11] = Wf, bf, Wft, bft -> multiplied by zero state, dead
    const float* Wz  = (const float*)d_in[12];  const float* bz  = (const float*)d_in[13];
    const float* Wo  = (const float*)d_in[14];  const float* bo  = (const float*)d_in[15];
    const float* Wd  = (const float*)d_in[16];  const float* bd  = (const float*)d_in[17];
    float* out = (float*)d_out;

    _Float16* xh = (_Float16*)d_ws;                                       // 32 MB
    _Float16* wp = (_Float16*)((char*)d_ws + (size_t)B_DIM * IN_DIM * 2); // 10 MB

    // pack x: 16.78M elems, 4 per thread
    clstm_pack_x<<<(B_DIM * IN_DIM / 4) / 256, 256, 0, stream>>>(x, xh);
    // pack 5 live weight halves: 5 * 2^20 halves
    clstm_pack_w<<<(NGATES << 20) / 256, 256, 0, stream>>>(wp, Wi, Wit, Wz, Wo, Wd);
    // fused GEMM + epilogue: 4 waves/block, each wave = 32x16 slab x 5 gates
    dim3 grid(B_DIM / 128, H_DIM / 16);
    clstm_gemm<<<grid, 128, 0, stream>>>(xh, wp, bi, bit, bz, bo, bd, t, last_t, out);
}
